// GRU_10471130268437
// MI455X (gfx1250) — matrix operand order repaired and checked by reading, not compiled, not run on hardware
//
#include <hip/hip_runtime.h>
#include <hip/hip_bf16.h>

// GRU: T=2048, B=64, I=H=256 -> output [T,B,H] f32 (+ hx echoed).
// Phase 0: convert x and x2h_w to f16 (one streaming pass).
// Phase 1: GI[T*B,768] = Xh @ x2h_w^T + x2h_b  (f16 WMMA, f32 acc, b128 loads).
// Phase 2: persistent scan, 16 WGs each owning a 16-col H slice:
//   TDM loads the step's GI tile to LDS, 12 wmma tiles compute gh,
//   gates in f32, h exchanged through L2 with async-to-LDS reload + sw grid barrier.

#define T_LEN 2048
#define B_SZ  64
#define I_SZ  256
#define H_SZ  256
#define H3    768
#define NWG   16
#define SCAN_THREADS 512

typedef __attribute__((ext_vector_type(16))) _Float16 v16h;
typedef __attribute__((ext_vector_type(8)))  _Float16 v8h;
typedef __attribute__((ext_vector_type(8)))  float    v8f;
typedef __attribute__((ext_vector_type(4)))  unsigned int v4u;
typedef __attribute__((ext_vector_type(8)))  int v8i;
typedef __attribute__((ext_vector_type(4)))  int v4i;

// Address-space-qualified vector pointers for the async global->LDS builtin.
typedef __attribute__((address_space(1))) v4i as1_v4i;
typedef __attribute__((address_space(3))) v4i as3_v4i;

#if defined(__HIP_DEVICE_COMPILE__) && __has_builtin(__builtin_amdgcn_global_load_async_to_lds_b128)
#define HAVE_ASYNC_LDS 1
#else
#define HAVE_ASYNC_LDS 0
#endif
#if defined(__HIP_DEVICE_COMPILE__) && __has_builtin(__builtin_amdgcn_tensor_load_to_lds) && \
    __has_builtin(__builtin_amdgcn_s_wait_tensorcnt)
#define HAVE_TDM 1
#else
#define HAVE_TDM 0
#endif

__device__ inline void wait_async0() {
#if __has_builtin(__builtin_amdgcn_s_wait_asynccnt)
  __builtin_amdgcn_s_wait_asynccnt(0);
#elif defined(__HIP_DEVICE_COMPILE__)
  asm volatile("s_wait_asynccnt 0" ::: "memory");
#endif
}

// ---------------- WMMA fragment helpers (16x16x32 f16, wave32) ----------------
// A 16(M)x32(K): lane = 16*lh + m; halves a[2v+e] = K (v<4?0:16)+2(v%4)+e+8lh
// -> two contiguous 8-half chunks at k0+8lh and k0+8lh+16.
__device__ inline v16h load_frag_A16(const _Float16* __restrict__ A, int lda,
                                     int m0, int k0) {
  const int lane = threadIdx.x & 31;
  const int lh   = lane >> 4;
  const _Float16* p = A + (m0 + (lane & 15)) * lda + k0 + 8 * lh;
  v8h lo = *(const v8h*)(p);
  v8h hi = *(const v8h*)(p + 16);
  return __builtin_shufflevector(lo, hi, 0, 1, 2, 3, 4, 5, 6, 7, 8, 9, 10, 11,
                                 12, 13, 14, 15);
}

// B 32(K)x16(N), B[k][n] = W[n0+n][k0+k]: lane = 16*lh + n; same K order as A
// -> two contiguous 8-half chunks at k0+8lh and k0+8lh+16.
__device__ inline v16h load_frag_B16(const _Float16* __restrict__ W, int ldw,
                                     int n0, int k0) {
  const int lane = threadIdx.x & 31;
  const int lh   = lane >> 4;
  const _Float16* p = W + (n0 + (lane & 15)) * ldw + k0 + 8 * lh;
  v8h lo = *(const v8h*)(p);
  v8h hi = *(const v8h*)(p + 16);
  return __builtin_shufflevector(lo, hi, 0, 1, 2, 3, 4, 5, 6, 7, 8, 9, 10, 11,
                                 12, 13, 14, 15);
}

__device__ inline v8f wmma_f16(v16h a, v16h b, v8f c) {
  return __builtin_amdgcn_wmma_f32_16x16x32_f16(false, a, false, b, (short)0, c,
                                                false, false);
}

// ---------------- Phase 0: f32 -> f16 conversion pass ----------------
__device__ inline void cvt8(const float* __restrict__ s, _Float16* __restrict__ d) {
  const float4 a = *(const float4*)(s);
  const float4 b = *(const float4*)(s + 4);
  v8h h;
  h[0] = (_Float16)a.x; h[1] = (_Float16)a.y; h[2] = (_Float16)a.z; h[3] = (_Float16)a.w;
  h[4] = (_Float16)b.x; h[5] = (_Float16)b.y; h[6] = (_Float16)b.z; h[7] = (_Float16)b.w;
  *(v8h*)(d) = h;
}

#define XCVT_BLOCKS 16384   // (T*B*I)/8/256
#define WCVT_BLOCKS 96      // (768*256)/8/256

__global__ __launch_bounds__(256) void gru_convert(const float* __restrict__ x,
                                                   const float* __restrict__ x2h_w,
                                                   _Float16* __restrict__ xh,
                                                   _Float16* __restrict__ wh) {
  if (blockIdx.x < XCVT_BLOCKS) {
    const size_t base = ((size_t)blockIdx.x * 256 + threadIdx.x) * 8;
    cvt8(x + base, xh + base);
  } else {
    const size_t base = ((size_t)(blockIdx.x - XCVT_BLOCKS) * 256 + threadIdx.x) * 8;
    cvt8(x2h_w + base, wh + base);
  }
}

// ---------------- Phase 1: GI = Xh @ x2h_w^T + x2h_b ----------------
// 8 waves/block; each wave one 16(M) x 64(N) strip (A fragment reused 4x).
__global__ __launch_bounds__(256) void gru_gi_gemm(
    const _Float16* __restrict__ xh,  // [T*B, 256] f16
    const _Float16* __restrict__ wh,  // [768, 256] f16
    const float* __restrict__ x2h_b,  // [768]
    float* __restrict__ gi)           // [T*B, 768]
{
  const int wtile = blockIdx.x * 8 + (threadIdx.x >> 5);
  const int m0    = (wtile / 12) * 16;
  const int n0    = (wtile % 12) * 64;

  v8f acc[4] = {};
#pragma unroll
  for (int kk = 0; kk < I_SZ / 32; ++kk) {
    const int k0 = kk * 32;
    v16h a = load_frag_A16(xh, I_SZ, m0, k0);
#pragma unroll
    for (int j = 0; j < 4; ++j) {
      v16h b = load_frag_B16(wh, I_SZ, n0 + j * 16, k0);
      acc[j] = wmma_f16(a, b, acc[j]);
    }
  }
  const int lane = threadIdx.x & 31;
  const int lh   = lane >> 4;
#pragma unroll
  for (int j = 0; j < 4; ++j) {
    const int n    = n0 + j * 16 + (lane & 15);
    const float bi = x2h_b[n];
#pragma unroll
    for (int r = 0; r < 8; ++r)
      gi[(size_t)(m0 + r + 8 * lh) * H3 + n] = acc[j][r] + bi;
  }
}

// ---------------- init: barrier reset, h0 f16 buffer, echo hx ----------------
__global__ void gru_init(const float* __restrict__ hx, _Float16* __restrict__ hbuf,
                         float* __restrict__ out_tail, int* __restrict__ bar) {
  const int i = blockIdx.x * blockDim.x + threadIdx.x;
  if (i < B_SZ * H_SZ) {
    const float v = hx[i];
    hbuf[i]     = (_Float16)v;
    out_tail[i] = v;
  }
  if (i == 0) { bar[0] = 0; bar[1] = 0; }
}

// ---------------- Phase 2: persistent recurrent scan ----------------
__global__ __launch_bounds__(SCAN_THREADS) void gru_scan(
    const float* __restrict__ gi,     // [T*B, 768]
    const float* __restrict__ hx,     // [B, 256]
    const float* __restrict__ h2h_w,  // [768, 256]
    const float* __restrict__ h2h_b,  // [768]
    float* __restrict__ out,          // [T, B, 256]
    _Float16* __restrict__ hbuf,      // [2, B*256] f16
    int* __restrict__ bar)            // [cnt, phase]
{
  __shared__ __align__(16) _Float16 h_half[B_SZ][H_SZ];  // 32 KB full h (f16)
  __shared__ __align__(16) _Float16 wsl[48][H_SZ];       // 24 KB weight slice
  __shared__ __align__(16) float    ghs[B_SZ][48];       // 12 KB gh slice
#if HAVE_TDM
  __shared__ __align__(16) float    gis[3][B_SZ][16];    // 12 KB GI tile via TDM
#endif
  __shared__ float hs[B_SZ][16];                         //  4 KB own h cols f32
  __shared__ float hb[48];

  const int tid = threadIdx.x;
  const int s   = blockIdx.x;               // H slice [16s, 16s+16)

  for (int idx = tid; idx < 48 * H_SZ; idx += SCAN_THREADS) {
    const int q = idx / H_SZ, k = idx % H_SZ;
    const int g = q >> 4, j = q & 15;
    wsl[q][k] = (_Float16)h2h_w[(g * H_SZ + s * 16 + j) * H_SZ + k];
  }
  if (tid < 48) hb[tid] = h2h_b[(tid >> 4) * H_SZ + s * 16 + (tid & 15)];
  for (int idx = tid; idx < B_SZ * H_SZ; idx += SCAN_THREADS)
    h_half[idx >> 8][idx & 255] = (_Float16)hx[idx];
  for (int idx = tid; idx < B_SZ * 16; idx += SCAN_THREADS)
    hs[idx >> 4][idx & 15] = hx[(idx >> 4) * H_SZ + s * 16 + (idx & 15)];
  __syncthreads();

  const int wave  = tid >> 5;
  const int item0 = tid * 2;                // 2 of 1024 (b,j) gate items

  for (int t = 0; t < T_LEN; ++t) {
#if !HAVE_TDM
    float gir[2], giz[2], gin[2];
#pragma unroll
    for (int u = 0; u < 2; ++u) {
      const int i = item0 + u;
      const float* grow =
          gi + (size_t)(t * B_SZ + (i >> 4)) * H3 + s * 16 + (i & 15);
      gir[u] = grow[0];
      giz[u] = grow[H_SZ];
      gin[u] = grow[2 * H_SZ];
    }
#endif
    if (t + 1 < T_LEN) {                    // warm L2 for next step's GI tile
      const float* nrow =
          gi + (size_t)((t + 1) * B_SZ + (item0 >> 4)) * H3 + s * 16 + (item0 & 15);
      __builtin_prefetch(nrow, 0, 1);
      __builtin_prefetch(nrow + H_SZ, 0, 1);
      __builtin_prefetch(nrow + 2 * H_SZ, 0, 1);
    }

    // GEMM: ghs[64][48] = h_half @ wsl^T ; waves 0..11 one 16x16 tile each.
    if (wave < 12) {
      const int m0 = (wave & 3) * 16;
      const int q0 = (wave >> 2) * 16;
      v8f acc = {};
#pragma unroll
      for (int kk = 0; kk < H_SZ / 32; ++kk) {
        v16h a = load_frag_A16(&h_half[0][0], H_SZ, m0, kk * 32);
        v16h b = load_frag_B16(&wsl[0][0], H_SZ, q0, kk * 32);
        acc = wmma_f16(a, b, acc);
      }
      const int lane = tid & 31;
      const int lh   = lane >> 4;
      const int q    = q0 + (lane & 15);
#pragma unroll
      for (int r = 0; r < 8; ++r) ghs[m0 + r + 8 * lh][q] = acc[r];
    }
#if HAVE_TDM
    else if (wave == 12) {
      // TDM: 3D tile x=16 f32, y=64 rows (stride 768), z=3 gates (stride 256)
      const unsigned long long ga =
          (unsigned long long)(const void*)(gi + (size_t)(t * B_SZ) * H3 + s * 16);
      const unsigned int lds_off =
          (unsigned int)(unsigned long long)(__attribute__((address_space(3)))
                                                 void*)&gis[0][0][0];
      v4u g0;
      g0[0] = 1u;                                   // count=1, user descriptor
      g0[1] = lds_off;                              // LDS byte address
      g0[2] = (unsigned int)(ga & 0xFFFFFFFFu);     // global addr lo
      g0[3] = (unsigned int)((ga >> 32) & 0x1FFFFFFu) | (2u << 30);  // hi | type=2
      v8i g1;
      g1[0] = (int)(2u << 16);                      // data_size = 4 bytes
      g1[1] = (int)(16u << 16);                     // tensor_dim0 = 16
      g1[2] = (int)(64u << 16);                     // tensor_dim1 = 64
      g1[3] = (int)(16u << 16);                     // tile_dim0 = 16
      g1[4] = (int)(64u | (3u << 16));              // tile_dim1=64, tile_dim2=3
      g1[5] = 768;                                  // tensor_dim0_stride
      g1[6] = (int)(256u << 16);                    // tensor_dim1_stride = 256
      g1[7] = 0;
      v4i g2 = {3, 0, 0, 0};                        // tensor_dim2 = 3
      v4i g3 = {0, 0, 0, 0};
      v8i g4 = {0, 0, 0, 0, 0, 0, 0, 0};            // extra group (6-arg form)
      __builtin_amdgcn_tensor_load_to_lds(g0, g1, g2, g3, g4, 0);
      __builtin_amdgcn_s_wait_tensorcnt(0);
    }
#endif
    __syncthreads();

    // Gate phase.
    const int nb = (t + 1) & 1;
    _Float16* hn = hbuf + nb * (B_SZ * H_SZ);
#pragma unroll
    for (int u = 0; u < 2; ++u) {
      const int i = item0 + u;
      const int b = i >> 4, j = i & 15;
#if HAVE_TDM
      const float gir_ = gis[0][b][j], giz_ = gis[1][b][j], gin_ = gis[2][b][j];
#else
      const float gir_ = gir[u], giz_ = giz[u], gin_ = gin[u];
#endif
      const float ghr = ghs[b][j]      + hb[j];
      const float ghz = ghs[b][16 + j] + hb[16 + j];
      const float ghn = ghs[b][32 + j] + hb[32 + j];
      const float r = 1.0f / (1.0f + __expf(-(gir_ + ghr)));
      const float z = 1.0f / (1.0f + __expf(-(giz_ + ghz)));
      const float n = tanhf(gin_ + r * ghn);
      const float hnew = (1.0f - z) * n + z * hs[b][j];
      out[(size_t)t * (B_SZ * H_SZ) + b * H_SZ + s * 16 + j] = hnew;
      hn[b * H_SZ + s * 16 + j] = (_Float16)hnew;
      hs[b][j] = hnew;
    }

    // Software grid barrier across the 16 persistent workgroups.
    __syncthreads();
    if (tid == 0) {
      __threadfence();
      const int arrived = atomicAdd(&bar[0], 1) + 1;
      if (arrived == NWG * (t + 1)) atomicExch(&bar[1], t + 1);
      while (atomicAdd(&bar[1], 0) < t + 1) __builtin_amdgcn_s_sleep(1);
      __threadfence();
    }
    __syncthreads();

    // Reload full updated h (32 KB) from L2 into LDS.
    {
      const char* srcb = (const char*)(hbuf + nb * (B_SZ * H_SZ));
      char* dstb = (char*)&h_half[0][0];
#pragma unroll
      for (int idx = tid; idx < (B_SZ * H_SZ * 2) / 16; idx += SCAN_THREADS) {
#if HAVE_ASYNC_LDS
        __builtin_amdgcn_global_load_async_to_lds_b128(
            (as1_v4i*)(srcb + idx * 16), (as3_v4i*)(dstb + idx * 16), 0, 0);
#else
        ((uint4*)dstb)[idx] = ((const uint4*)srcb)[idx];
#endif
      }
#if HAVE_ASYNC_LDS
      wait_async0();
#endif
    }
    __syncthreads();
  }
}

extern "C" void kernel_launch(void* const* d_in, const int* in_sizes, int n_in,
                              void* d_out, int out_size, void* d_ws, size_t ws_size,
                              hipStream_t stream) {
  const float* x     = (const float*)d_in[0];
  const float* hx    = (const float*)d_in[1];
  const float* x2h_w = (const float*)d_in[2];
  const float* h2h_w = (const float*)d_in[3];
  const float* x2h_b = (const float*)d_in[4];
  const float* h2h_b = (const float*)d_in[5];
  float* out = (float*)d_out;

  char* ws = (char*)d_ws;
  float* gi = (float*)ws;                                         // 402.7 MB
  const size_t gi_bytes = (size_t)T_LEN * B_SZ * H3 * sizeof(float);
  _Float16* xh = (_Float16*)(ws + gi_bytes);                      // 67.1 MB
  const size_t xh_bytes = (size_t)T_LEN * B_SZ * I_SZ * sizeof(_Float16);
  _Float16* wh = (_Float16*)(ws + gi_bytes + xh_bytes);           // 384 KB
  const size_t wh_bytes = (size_t)H3 * I_SZ * sizeof(_Float16);
  _Float16* hbuf = (_Float16*)(ws + gi_bytes + xh_bytes + wh_bytes);  // 64 KB
  int* bar = (int*)(ws + gi_bytes + xh_bytes + wh_bytes +
                    2 * B_SZ * H_SZ * sizeof(_Float16));

  gru_init<<<64, 256, 0, stream>>>(hx, hbuf, out + (size_t)T_LEN * B_SZ * H_SZ, bar);
  gru_convert<<<XCVT_BLOCKS + WCVT_BLOCKS, 256, 0, stream>>>(x, x2h_w, xh, wh);
  gru_gi_gemm<<<12288, 256, 0, stream>>>(xh, wh, x2h_b, gi);
  gru_scan<<<NWG, SCAN_THREADS, 0, stream>>>(gi, hx, h2h_w, h2h_b, out, hbuf, bar);
}
